// OnlineMemoryAttention_79663053406293
// MI455X (gfx1250) — matrix operand order repaired and checked by reading, not compiled, not run on hardware
//
#include <hip/hip_runtime.h>
#include <math.h>

// ---------------------------------------------------------------------------
// OnlineMemoryAttention for MI455X (gfx1250, wave32, WMMA + TDM).
// Big GEMMs + flash attention use v_wmma_f32_16x16x32_bf16 (fp32 accum).
// GEMM tiles are fed to LDS by the Tensor Data Mover (tensor_load_to_lds,
// TENSORcnt double buffering). Rank-6 incidence GEMM uses
// v_wmma_f32_16x16x4_f32 (full fp32, K=6 padded to 8).
// ---------------------------------------------------------------------------

#define DEV static __device__ __forceinline__

typedef __attribute__((ext_vector_type(16))) __bf16 v16bf;
typedef __attribute__((ext_vector_type(8)))  __bf16 v8bf;
typedef __attribute__((ext_vector_type(8)))  float  v8f;
typedef __attribute__((ext_vector_type(2)))  float  v2f;
typedef __attribute__((ext_vector_type(4)))  unsigned v4u;
typedef __attribute__((ext_vector_type(8)))  int    v8i;
typedef __attribute__((ext_vector_type(4)))  int    v4i;

static constexpr int B  = 2;
static constexpr int T  = 1536;
static constexpr int D  = 1024;
static constexpr int H  = 16;
static constexpr int DH = 64;
static constexpr int BT = B * T;

DEV v8bf ld8(const __bf16* p) { return *reinterpret_cast<const v8bf*>(p); }
DEV v16bf cat16(v8bf lo, v8bf hi) {
  return __builtin_shufflevector(lo, hi, 0,1,2,3,4,5,6,7,8,9,10,11,12,13,14,15);
}

// ---------------------------------------------------------------------------
// TDM: post a 2-D tile load (tile_dim = 64 rows x 32 cols of 2-byte elements)
// from a row-major tensor (rows x d0elems, line stride strideElems) into LDS.
// D# layout per CDNA5 ISA ch.8: group0 = {count/type/addr}, group1 = dims.
// Issued by one wave; completion tracked with TENSORcnt.
// ---------------------------------------------------------------------------
DEV void tdm_load_tile64x32_bf16(unsigned lds_off, const __bf16* gptr,
                                 unsigned d0elems, unsigned rows, unsigned strideElems) {
  unsigned long long ga = (unsigned long long)(uintptr_t)gptr;
  v4u g0;
  g0[0] = 1u;                                             // count=1, user desc
  g0[1] = lds_off;                                        // lds_addr (bytes)
  g0[2] = (unsigned)(ga & 0xFFFFFFFFull);                 // global_addr[31:0]
  g0[3] = (unsigned)((ga >> 32) & 0x1FFFFFFull) | (2u << 30);  // addr[56:32] | type=2
  v8i g1;
  g1[0] = (int)(1u << 16);                                // data_size=1 -> 2 bytes
  g1[1] = (int)((d0elems & 0xFFFFu) << 16);               // tensor_dim0[15:0]
  g1[2] = (int)(((d0elems >> 16) & 0xFFFFu) | ((rows & 0xFFFFu) << 16));  // dim0 hi | dim1 lo
  g1[3] = (int)(((rows >> 16) & 0xFFFFu) | (32u << 16));  // dim1 hi | tile_dim0=32
  g1[4] = 64;                                             // tile_dim1=64, tile_dim2=0
  g1[5] = (int)strideElems;                               // tensor_dim0_stride[31:0]
  g1[6] = 0;                                              // stride hi | dim1_stride lo
  g1[7] = 0;
  v4i gz4 = {0, 0, 0, 0};
  v8i gz8 = {0, 0, 0, 0, 0, 0, 0, 0};
  // 6-arg form (clang-23 / therock-10.0 headers): g0, g1, g2, g3, g4, cpol
  __builtin_amdgcn_tensor_load_to_lds(g0, g1, gz4, gz4, gz8, 0);
}

// ---------------------------------------------------------------------------
// fp32 -> bf16 convert
// ---------------------------------------------------------------------------
__global__ void k_f2bf(const float* __restrict__ src, __bf16* __restrict__ dst, int n) {
  int i = blockIdx.x * blockDim.x + threadIdx.x;
  if (i < n) dst[i] = (__bf16)src[i];
}

// w stored (K, N) row-major (torch transposed Linear weight); emit wt (N, K) bf16
__global__ void k_wT2bf(const float* __restrict__ w, __bf16* __restrict__ wt, int K, int N) {
  int i = blockIdx.x * blockDim.x + threadIdx.x;
  if (i < K * N) {
    int n = i / K, k = i % K;
    wt[i] = (__bf16)w[(size_t)k * N + n];
  }
}

// ---------------------------------------------------------------------------
// bf16 GEMM: C(M,N) = A(M,K) * Bt(N,K)^T + bias, fp32 accumulate.
// Block = 256 threads = 8 waves; block tile 64x64. Wave w: rows (w&3)*16,
// cols (w>>2)*32 (two 16x16 subtiles). A/B K-slabs (64x32 bf16 = 4KB each)
// are DMA'd into double-buffered LDS by wave 0 via the Tensor Data Mover;
// waves consume them with ds_load_b128 fragment reads.
// A fragment: lane L row (L&15); K = {h*8..h*8+7} u {16+h*8..} (h = L>>4)
// B fragment: lane L col (L&15); K = {h*8..h*8+7} u {16+h*8..} (h = L>>4)
// C fragment: lane L col (L&15), rows h*8 + j
// ---------------------------------------------------------------------------
template <bool BF16OUT>
__global__ void k_gemm(const __bf16* __restrict__ A, const __bf16* __restrict__ Bt,
                       const float* __restrict__ bias, void* __restrict__ Cout,
                       int M, int N, int K) {
  __shared__ alignas(16) __bf16 sA[2][64 * 32];
  __shared__ alignas(16) __bf16 sB[2][64 * 32];
  const int lane = threadIdx.x & 31;
  const int w    = threadIdx.x >> 5;
  const int half = lane >> 4, r = lane & 15;
  const int wm = w & 3, wn = w >> 2;
  const int m0 = blockIdx.x * 64;
  const int n0 = blockIdx.y * 64;
  const bool loader = (w == 0);

  const __bf16* Abase = A  + (size_t)m0 * K;
  const __bf16* Bbase = Bt + (size_t)n0 * K;
  const int nk = K / 32;

  if (loader) {
    tdm_load_tile64x32_bf16((unsigned)(uintptr_t)&sA[0][0], Abase, (unsigned)K, (unsigned)M, (unsigned)K);
    tdm_load_tile64x32_bf16((unsigned)(uintptr_t)&sB[0][0], Bbase, (unsigned)K, (unsigned)N, (unsigned)K);
  }

  v8f acc[2] = {v8f{}, v8f{}};
  for (int kt = 0; kt < nk; ++kt) {
    const int cur = kt & 1;
    if (loader) {
      if (kt + 1 < nk) {  // prefetch next slab, then wait for current pair
        tdm_load_tile64x32_bf16((unsigned)(uintptr_t)&sA[cur ^ 1][0],
                                Abase + (kt + 1) * 32, (unsigned)K, (unsigned)M, (unsigned)K);
        tdm_load_tile64x32_bf16((unsigned)(uintptr_t)&sB[cur ^ 1][0],
                                Bbase + (kt + 1) * 32, (unsigned)K, (unsigned)N, (unsigned)K);
        __builtin_amdgcn_s_wait_tensorcnt(2);
      } else {
        __builtin_amdgcn_s_wait_tensorcnt(0);
      }
    }
    __syncthreads();  // current slab visible to all waves

    const __bf16* a0 = &sA[cur][(wm * 16 + r) * 32];
    v16bf af = cat16(ld8(a0 + half * 8), ld8(a0 + 16 + half * 8));
#pragma unroll
    for (int s = 0; s < 2; ++s) {
      const __bf16* b0 = &sB[cur][(wn * 32 + s * 16 + r) * 32];
      v16bf bf = cat16(ld8(b0 + half * 8), ld8(b0 + 16 + half * 8));
      acc[s] = __builtin_amdgcn_wmma_f32_16x16x32_bf16(false, af, false, bf,
                                                       (short)0, acc[s], false, false);
    }
    __syncthreads();  // all reads done before the next DMA overwrites this buffer
  }

#pragma unroll
  for (int s = 0; s < 2; ++s) {
    const int nsub = n0 + wn * 32 + s * 16;
    if (nsub >= N) continue;  // wave-uniform (N is a multiple of 16)
    const float bv = bias ? bias[nsub + r] : 0.0f;
#pragma unroll
    for (int j = 0; j < 8; ++j) {
      const int row = m0 + wm * 16 + half * 8 + j;
      const float v = acc[s][j] + bv;
      if (BF16OUT)
        ((__bf16*)Cout)[(size_t)row * N + nsub + r] = (__bf16)v;
      else
        ((float*)Cout)[(size_t)row * N + nsub + r] = v;
    }
  }
}

// ---------------------------------------------------------------------------
// Repack V from Yqkv (BT, 3D) into vT (B*H, DH, T) so P@V B-fragments load
// contiguously along s.
// ---------------------------------------------------------------------------
__global__ void k_packVT(const __bf16* __restrict__ Yqkv, __bf16* __restrict__ vT) {
  int i = blockIdx.x * blockDim.x + threadIdx.x;
  if (i >= BT * D) return;
  int bt = i / D, c = i % D;
  int b = bt / T, t = bt % T;
  int h = c / DH, d = c % DH;
  vT[((size_t)(b * H + h) * DH + d) * T + t] = Yqkv[(size_t)bt * (3 * D) + 2 * D + c];
}

// ---------------------------------------------------------------------------
// Exterior algebra: Plucker lines, J-transform, seed quadratic form.
// ---------------------------------------------------------------------------
DEV void ext6(const float* a, const float* b, float* L) {
  L[0] = a[0] * b[1] - a[1] * b[0];
  L[1] = a[0] * b[2] - a[2] * b[0];
  L[2] = a[0] * b[3] - a[3] * b[0];
  L[3] = a[1] * b[2] - a[2] * b[1];
  L[4] = a[1] * b[3] - a[3] * b[1];
  L[5] = a[2] * b[3] - a[3] * b[2];
  float n = 0.f;
#pragma unroll
  for (int q = 0; q < 6; ++q) n += L[q] * L[q];
  n = fmaxf(sqrtf(n), 1e-12f);
#pragma unroll
  for (int q = 0; q < 6; ++q) L[q] /= n;
}
DEV void jmul(const float* L, float* J) {  // v @ J6 = (v5, -v4, v3, v2, -v1, v0)
  J[0] = L[5]; J[1] = -L[4]; J[2] = L[3]; J[3] = L[2]; J[4] = -L[1]; J[5] = L[0];
}

__global__ void k_lines(const float* __restrict__ p1w, const float* __restrict__ p2w,
                        const float* __restrict__ p1r, const float* __restrict__ p2r,
                        const float* __restrict__ gram,
                        float* __restrict__ rl, float* __restrict__ jw,
                        float* __restrict__ seedterm) {
  int i = blockIdx.x * blockDim.x + threadIdx.x;  // over B*T*H
  if (i >= BT * H) return;
  int h = i % H, bt = i / H;
  int b = bt / T, t = bt % T;
  int bh = b * H + h;

  float a[4], c[4], L[6], J[6];
  // write line: x_prev projection = row (bt-1) of x@w1_write (zeros at t==0)
#pragma unroll
  for (int q = 0; q < 4; ++q) {
    a[q] = (t > 0) ? p1w[(size_t)(bt - 1) * (4 * H) + h * 4 + q] : 0.f;
    c[q] = p2w[(size_t)bt * (4 * H) + h * 4 + q];
  }
  ext6(a, c, L);
  jmul(L, J);
#pragma unroll
  for (int q = 0; q < 6; ++q) jw[((size_t)bh * T + t) * 6 + q] = J[q];

  // read line
#pragma unroll
  for (int q = 0; q < 4; ++q) {
    a[q] = p1r[(size_t)bt * (4 * H) + h * 4 + q];
    c[q] = p2r[(size_t)bt * (4 * H) + h * 4 + q];
  }
  ext6(a, c, L);
#pragma unroll
  for (int q = 0; q < 6; ++q) rl[((size_t)bh * T + t) * 6 + q] = L[q];
  jmul(L, J);

  const float* G = gram + (size_t)bh * 36;
  float s = 0.f;
#pragma unroll
  for (int qi = 0; qi < 6; ++qi) {
    float rowdot = 0.f;
#pragma unroll
    for (int qj = 0; qj < 6; ++qj) rowdot += G[qi * 6 + qj] * J[qj];
    s += J[qi] * rowdot;
  }
  seedterm[(size_t)bh * T + t] = s;
}

// ---------------------------------------------------------------------------
// Flash attention, one wave (32 threads) per (b,h, 16 query rows).
// Q/K read straight out of Yqkv (row stride 3D); V via vT.
// ---------------------------------------------------------------------------
__global__ void k_attn(const __bf16* __restrict__ Yqkv, const __bf16* __restrict__ vT,
                       float* __restrict__ seq_out) {
  const int lane = threadIdx.x & 31;
  const int bh = blockIdx.x;
  const int b = bh / H, h = bh % H;
  const int tb = blockIdx.y;
  const int half = lane >> 4, r = lane & 15;
  const float scale = 0.125f;  // dh^-0.5

  // Q A-fragments (row = tb*16 + r, K = d, 2 frags cover dh=64)
  const __bf16* qbase = Yqkv + (size_t)(b * T + tb * 16 + r) * (3 * D) + h * DH;
  v16bf qf[2];
#pragma unroll
  for (int f = 0; f < 2; ++f)
    qf[f] = cat16(ld8(qbase + f * 32 + half * 8), ld8(qbase + f * 32 + 16 + half * 8));

  v8f acc[4] = {v8f{}, v8f{}, v8f{}, v8f{}};
  float m[8], l[8];
#pragma unroll
  for (int j = 0; j < 8; ++j) { m[j] = -1e30f; l[j] = 0.f; }

  __shared__ alignas(16) __bf16 ldsP[16 * 32];

  const int s_end = tb * 16 + 16;
  for (int s0 = 0; s0 < s_end; s0 += 32) {
    // ---- S = Q K^T for a 16x32 tile of s
    v8f S[2];
#pragma unroll
    for (int nt = 0; nt < 2; ++nt) {
      const int scol = s0 + nt * 16 + r;
      const __bf16* kbase = Yqkv + (size_t)(b * T + scol) * (3 * D) + D + h * DH;
      v8f c = {};
#pragma unroll
      for (int f = 0; f < 2; ++f) {
        v16bf kf = cat16(ld8(kbase + f * 32 + half * 8),
                         ld8(kbase + f * 32 + 16 + half * 8));
        c = __builtin_amdgcn_wmma_f32_16x16x32_bf16(false, qf[f], false, kf,
                                                    (short)0, c, false, false);
      }
      S[nt] = c;
    }

    // ---- scale + causal mask + row max (rows live on 8 j's per lane)
    float mnew[8];
#pragma unroll
    for (int j = 0; j < 8; ++j) {
      const int t = tb * 16 + half * 8 + j;
#pragma unroll
      for (int nt = 0; nt < 2; ++nt) {
        const int s = s0 + nt * 16 + r;
        float v = S[nt][j] * scale;
        S[nt][j] = (s <= t) ? v : -1e30f;
      }
      float mm = fmaxf(S[0][j], S[1][j]);
      mm = fmaxf(mm, __shfl_xor(mm, 1, 32));
      mm = fmaxf(mm, __shfl_xor(mm, 2, 32));
      mm = fmaxf(mm, __shfl_xor(mm, 4, 32));
      mm = fmaxf(mm, __shfl_xor(mm, 8, 32));
      mnew[j] = mm;
    }

    // ---- online softmax update; stage P (C-layout) into LDS as bf16
    float corr[8];
#pragma unroll
    for (int j = 0; j < 8; ++j) {
      const float mu = fmaxf(m[j], mnew[j]);
      corr[j] = __expf(m[j] - mu);
      const float p0 = __expf(S[0][j] - mu);
      const float p1 = __expf(S[1][j] - mu);
      const int row = half * 8 + j;
      ldsP[row * 32 + r] = (__bf16)p0;
      ldsP[row * 32 + 16 + r] = (__bf16)p1;
      float ps = p0 + p1;
      ps += __shfl_xor(ps, 1, 32);
      ps += __shfl_xor(ps, 2, 32);
      ps += __shfl_xor(ps, 4, 32);
      ps += __shfl_xor(ps, 8, 32);
      m[j] = mu;
      l[j] = l[j] * corr[j] + ps;
    }
    asm volatile("s_wait_dscnt 0" ::: "memory");  // stores visible before reload

    // ---- reload P in A-fragment layout (row = r, K = s_local)
    v16bf pf = cat16(*reinterpret_cast<const v8bf*>(&ldsP[r * 32 + half * 8]),
                     *reinterpret_cast<const v8bf*>(&ldsP[r * 32 + 16 + half * 8]));

    // ---- acc = acc*corr + P @ V  (4 n-tiles of d)
#pragma unroll
    for (int nt = 0; nt < 4; ++nt) {
#pragma unroll
      for (int j = 0; j < 8; ++j) acc[nt][j] *= corr[j];
      const __bf16* vbase = vT + ((size_t)bh * DH + nt * 16 + r) * T + s0;
      v16bf vf = cat16(ld8(vbase + half * 8), ld8(vbase + 16 + half * 8));
      acc[nt] = __builtin_amdgcn_wmma_f32_16x16x32_bf16(false, pf, false, vf,
                                                        (short)0, acc[nt], false, false);
    }
  }

#pragma unroll
  for (int nt = 0; nt < 4; ++nt)
#pragma unroll
    for (int j = 0; j < 8; ++j) {
      const int t = tb * 16 + half * 8 + j;
      seq_out[((size_t)b * T + t) * D + h * DH + nt * 16 + r] = acc[nt][j] / l[j];
    }
}

// ---------------------------------------------------------------------------
// Incidence: inc(t,s) = <read_t, Jwrite_s> (K=6 pad 8) via f32 WMMA, then
// mem contribution = sum_{s<t} decay^(t-s) inc^2. One wave per (b,h,16 rows).
// ---------------------------------------------------------------------------
__global__ void k_incidence(const float* __restrict__ rl, const float* __restrict__ jw,
                            float* __restrict__ incsum) {
  const int lane = threadIdx.x & 31;
  const int bh = blockIdx.x;
  const int t0 = blockIdx.y * 16;
  const int half = lane >> 4, r = lane & 15;

  v2f a[2];
#pragma unroll
  for (int g = 0; g < 2; ++g)
#pragma unroll
    for (int i = 0; i < 2; ++i) {
      const int k = g * 4 + half * 2 + i;
      a[g][i] = (k < 6) ? rl[((size_t)bh * T + t0 + r) * 6 + k] : 0.f;
    }

  float wsum[8] = {0, 0, 0, 0, 0, 0, 0, 0};
  const float LOGD = -0.010050335853501441f;  // ln(0.99)

  for (int s0 = 0; s0 <= t0; s0 += 16) {
    const int scol = s0 + r;
    v2f bb[2];
#pragma unroll
    for (int g = 0; g < 2; ++g)
#pragma unroll
      for (int i = 0; i < 2; ++i) {
        const int k = g * 4 + half * 2 + i;
        bb[g][i] = (k < 6) ? jw[((size_t)bh * T + scol) * 6 + k] : 0.f;
      }
    v8f c = {};
    c = __builtin_amdgcn_wmma_f32_16x16x4_f32(false, a[0], false, bb[0], (short)0, c, false, false);
    c = __builtin_amdgcn_wmma_f32_16x16x4_f32(false, a[1], false, bb[1], (short)0, c, false, false);
#pragma unroll
    for (int j = 0; j < 8; ++j) {
      const int t = t0 + half * 8 + j;
      const int ds = t - scol;
      const float wgt = (ds > 0) ? __expf((float)ds * LOGD) : 0.f;
      wsum[j] += wgt * c[j] * c[j];
    }
  }
#pragma unroll
  for (int j = 0; j < 8; ++j) {
    float s = wsum[j];
    s += __shfl_xor(s, 1, 32);
    s += __shfl_xor(s, 2, 32);
    s += __shfl_xor(s, 4, 32);
    s += __shfl_xor(s, 8, 32);
    if (r == 0) incsum[(size_t)bh * T + t0 + half * 8 + j] = s;
  }
}

// ---------------------------------------------------------------------------
// Gating + combine
// ---------------------------------------------------------------------------
__global__ void k_gate(const float* __restrict__ incsum, const float* __restrict__ seedterm,
                       const float* __restrict__ gate_lin, const float* __restrict__ mem_scale,
                       const float* __restrict__ seed_scale, float* __restrict__ gated) {
  int bt = blockIdx.x * blockDim.x + threadIdx.x;
  if (bt >= BT) return;
  int b = bt / T, t = bt % T;
  float acc = 0.f;
#pragma unroll
  for (int h = 0; h < H; ++h) {
    const int bh = b * H + h;
    const float ms = incsum[(size_t)bh * T + t] + seed_scale[h] * seedterm[(size_t)bh * T + t];
    const float g1 = 1.f / (1.f + __expf(-ms * mem_scale[h]));
    const float g2 = 1.f / (1.f + __expf(-gate_lin[(size_t)bt * H + h]));
    acc += g1 * g2;
  }
  gated[bt] = acc / (float)H;
}

__global__ void k_z(const float* __restrict__ seq_out, const float* __restrict__ mem_val,
                    const float* __restrict__ gated, __bf16* __restrict__ zb) {
  int i = blockIdx.x * blockDim.x + threadIdx.x;
  if (i >= BT * D) return;
  zb[i] = (__bf16)(seq_out[i] + gated[i / D] * mem_val[i]);
}

// ---------------------------------------------------------------------------
// Orchestration
// ---------------------------------------------------------------------------
extern "C" void kernel_launch(void* const* d_in, const int* in_sizes, int n_in,
                              void* d_out, int out_size, void* d_ws, size_t ws_size,
                              hipStream_t stream) {
  (void)in_sizes; (void)n_in; (void)out_size; (void)ws_size;
  const float* x          = (const float*)d_in[0];
  const float* gram       = (const float*)d_in[1];
  const float* qkv_w      = (const float*)d_in[2];
  const float* qkv_b      = (const float*)d_in[3];
  const float* w1_write   = (const float*)d_in[4];
  const float* w2_write   = (const float*)d_in[5];
  const float* w1_read    = (const float*)d_in[6];
  const float* w2_read    = (const float*)d_in[7];
  const float* memv_w     = (const float*)d_in[8];
  const float* memv_b     = (const float*)d_in[9];
  const float* memg_w     = (const float*)d_in[10];
  const float* memg_b     = (const float*)d_in[11];
  const float* mem_scale  = (const float*)d_in[12];
  const float* seed_scale = (const float*)d_in[13];
  const float* out_w      = (const float*)d_in[14];
  const float* out_b      = (const float*)d_in[15];

  char* p = (char*)d_ws;
  auto alloc = [&](size_t bytes) -> char* {
    char* ret = p;
    p += (bytes + 255) & ~(size_t)255;
    return ret;
  };
  __bf16* xb      = (__bf16*)alloc((size_t)BT * D * 2);
  __bf16* qkvwT   = (__bf16*)alloc((size_t)3 * D * D * 2);
  __bf16* memvwT  = (__bf16*)alloc((size_t)D * D * 2);
  __bf16* memgwT  = (__bf16*)alloc((size_t)H * D * 2);
  __bf16* w1wT    = (__bf16*)alloc((size_t)4 * H * D * 2);
  __bf16* w2wT    = (__bf16*)alloc((size_t)4 * H * D * 2);
  __bf16* w1rT    = (__bf16*)alloc((size_t)4 * H * D * 2);
  __bf16* w2rT    = (__bf16*)alloc((size_t)4 * H * D * 2);
  __bf16* outwT   = (__bf16*)alloc((size_t)D * D * 2);
  __bf16* Yqkv    = (__bf16*)alloc((size_t)BT * 3 * D * 2);
  __bf16* vTb     = (__bf16*)alloc((size_t)BT * D * 2);
  float*  memval  = (float*)alloc((size_t)BT * D * 4);
  float*  gatelin = (float*)alloc((size_t)BT * H * 4);
  float*  p1w     = (float*)alloc((size_t)BT * 4 * H * 4);
  float*  p2w     = (float*)alloc((size_t)BT * 4 * H * 4);
  float*  p1r     = (float*)alloc((size_t)BT * 4 * H * 4);
  float*  p2r     = (float*)alloc((size_t)BT * 4 * H * 4);
  float*  rl      = (float*)alloc((size_t)B * H * T * 6 * 4);
  float*  jwb     = (float*)alloc((size_t)B * H * T * 6 * 4);
  float*  seedt   = (float*)alloc((size_t)B * H * T * 4);
  float*  incsum  = (float*)alloc((size_t)B * H * T * 4);
  float*  seqout  = (float*)alloc((size_t)BT * D * 4);
  float*  gated   = (float*)alloc((size_t)BT * 4);
  __bf16* zb      = (__bf16*)alloc((size_t)BT * D * 2);

  const int blk = 256;
  // 1) convert x
  k_f2bf<<<(BT * D + blk - 1) / blk, blk, 0, stream>>>(x, xb, BT * D);
  // 2) transpose+convert weights
  k_wT2bf<<<(D * 3 * D + blk - 1) / blk, blk, 0, stream>>>(qkv_w, qkvwT, D, 3 * D);
  k_wT2bf<<<(D * D + blk - 1) / blk, blk, 0, stream>>>(memv_w, memvwT, D, D);
  k_wT2bf<<<(D * H + blk - 1) / blk, blk, 0, stream>>>(memg_w, memgwT, D, H);
  k_wT2bf<<<(D * 4 * H + blk - 1) / blk, blk, 0, stream>>>(w1_write, w1wT, D, 4 * H);
  k_wT2bf<<<(D * 4 * H + blk - 1) / blk, blk, 0, stream>>>(w2_write, w2wT, D, 4 * H);
  k_wT2bf<<<(D * 4 * H + blk - 1) / blk, blk, 0, stream>>>(w1_read, w1rT, D, 4 * H);
  k_wT2bf<<<(D * 4 * H + blk - 1) / blk, blk, 0, stream>>>(w2_read, w2rT, D, 4 * H);
  k_wT2bf<<<(D * D + blk - 1) / blk, blk, 0, stream>>>(out_w, outwT, D, D);

  // 3) projections (TDM-fed WMMA bf16 GEMMs; block tile 64x64)
  k_gemm<true><<<dim3(BT / 64, (3 * D) / 64), blk, 0, stream>>>(xb, qkvwT, qkv_b, Yqkv, BT, 3 * D, D);
  k_gemm<false><<<dim3(BT / 64, D / 64), blk, 0, stream>>>(xb, memvwT, memv_b, memval, BT, D, D);
  k_gemm<false><<<dim3(BT / 64, 1), blk, 0, stream>>>(xb, memgwT, memg_b, gatelin, BT, H, D);
  k_gemm<false><<<dim3(BT / 64, 1), blk, 0, stream>>>(xb, w1wT, nullptr, p1w, BT, 4 * H, D);
  k_gemm<false><<<dim3(BT / 64, 1), blk, 0, stream>>>(xb, w2wT, nullptr, p2w, BT, 4 * H, D);
  k_gemm<false><<<dim3(BT / 64, 1), blk, 0, stream>>>(xb, w1rT, nullptr, p1r, BT, 4 * H, D);
  k_gemm<false><<<dim3(BT / 64, 1), blk, 0, stream>>>(xb, w2rT, nullptr, p2r, BT, 4 * H, D);

  // 4) V transpose repack, exterior lines
  k_packVT<<<(BT * D + blk - 1) / blk, blk, 0, stream>>>(Yqkv, vTb);
  k_lines<<<(BT * H + blk - 1) / blk, blk, 0, stream>>>(p1w, p2w, p1r, p2r, gram, rl, jwb, seedt);

  // 5) flash attention + incidence
  k_attn<<<dim3(B * H, T / 16), 32, 0, stream>>>(Yqkv, vTb, seqout);
  k_incidence<<<dim3(B * H, T / 16), 32, 0, stream>>>(rl, jwb, incsum);

  // 6) gate, combine, output projection
  k_gate<<<(BT + blk - 1) / blk, blk, 0, stream>>>(incsum, seedt, gatelin, mem_scale, seed_scale, gated);
  k_z<<<(BT * D + blk - 1) / blk, blk, 0, stream>>>(seqout, memval, gated, zb);
  k_gemm<false><<<dim3(BT / 64, D / 64), blk, 0, stream>>>(zb, outwT, out_b, (float*)d_out, BT, D, D);
}
